// LeakyAttention_83167746720058
// MI455X (gfx1250) — matrix operand order repaired and checked by reading, not compiled, not run on hardware
//
#include <hip/hip_runtime.h>
#include <hip/hip_bf16.h>
#include <math.h>

// ---------------------------------------------------------------------------
// Problem constants (match reference)
// ---------------------------------------------------------------------------
#define BB   32          // batch
#define CC   128         // channels / d_model
#define HH   110         // spatial
#define SS   (HH*HH)     // 12100 pixels
#define KT   768         // tokens kept
#define NHH  8           // heads
#define HD   16          // head dim
#define MR   (BB*KT)     // 24576 token rows

typedef __attribute__((ext_vector_type(8)))  _Float16 v8h;
typedef __attribute__((ext_vector_type(16))) _Float16 v16h;
typedef __attribute__((ext_vector_type(8)))  float    v8f;

// ---------------------------------------------------------------------------
// WMMA fragment helpers (wave32; contiguous k order chosen here for A/B, C/D per ISA)
//  A (16x32 f16): lane = (half, m);  a[i]   = A[m][half*16 + i]
//  B (32x16 f16): lane = (half, n);  b[i]   = B[half*16 + i][n]
//  C/D (16x16 f32): lane = (half, n); d[r]  = D[half*8 + r][n]
// ---------------------------------------------------------------------------
__device__ __forceinline__ v16h load_a_frag(const _Float16* row, int half_) {
  return *(const v16h*)(row + half_ * 16);
}
__device__ __forceinline__ v16h load_b_frag(const _Float16* col, int half_) {
  return *(const v16h*)(col + half_ * 16);
}
__device__ __forceinline__ v8f wmma_f16(v16h a, v16h b, v8f c) {
  return __builtin_amdgcn_wmma_f32_16x16x32_f16(false, a, false, b, (short)0, c,
                                                false, false);
}

// ---------------------------------------------------------------------------
// 1) Dilated conv (k=7, d=5, p=15) + BN(eval) + exact GELU + sigmoid -> imp
// ---------------------------------------------------------------------------
__global__ __launch_bounds__(256)
void conv_imp_kernel(const float* __restrict__ x, const float* __restrict__ w,
                     const float* __restrict__ cb, const float* __restrict__ bng,
                     const float* __restrict__ bnb, const float* __restrict__ bnm,
                     const float* __restrict__ bnv, float* __restrict__ imp) {
  int p = blockIdx.x * 256 + threadIdx.x;
  int b = blockIdx.y;
  if (p >= SS) return;
  int oy = p / HH, ox = p - oy * HH;
  float acc = 0.f;
  for (int c = 0; c < CC; ++c) {
    const float* xc = x + ((size_t)b * CC + c) * SS;
    const float* wc = w + c * 49;
#pragma unroll
    for (int ky = 0; ky < 7; ++ky) {
      int iy = oy - 15 + 5 * ky;
      if ((unsigned)iy < (unsigned)HH) {
        const float* xr = xc + iy * HH;
#pragma unroll
        for (int kx = 0; kx < 7; ++kx) {
          int ix = ox - 15 + 5 * kx;
          if ((unsigned)ix < (unsigned)HH) acc = fmaf(xr[ix], wc[ky * 7 + kx], acc);
        }
      }
    }
  }
  acc += cb[0];
  float sc = bng[0] * rsqrtf(bnv[0] + 1e-5f);
  float y  = (acc - bnm[0]) * sc + bnb[0];
  float g  = 0.5f * y * (1.f + erff(y * 0.70710678118654752f)); // exact GELU
  imp[(size_t)b * SS + p] = 1.f / (1.f + __expf(-g));
}

// 2) default output: out[b,c,s] = x[b,c,s] * imp[b,s]
__global__ __launch_bounds__(256)
void mulimp_kernel(const float* __restrict__ x, const float* __restrict__ imp,
                   float* __restrict__ out) {
  size_t i = (size_t)blockIdx.x * 256 + threadIdx.x;
  if (i >= (size_t)BB * CC * SS) return;
  size_t b = i / ((size_t)CC * SS);
  size_t s = i % SS;
  out[i] = x[i] * imp[b * SS + s];
}

// ---------------------------------------------------------------------------
// 3) top-K radix select on positive float bits (one block of 256 per batch)
// ---------------------------------------------------------------------------
__global__ __launch_bounds__(256)
void topk_kernel(const float* __restrict__ imp, int* __restrict__ idx) {
  int b = blockIdx.x, tid = threadIdx.x;
  const float* v = imp + (size_t)b * SS;
  __shared__ unsigned hist[256];
  __shared__ unsigned sh_pref;
  __shared__ int sh_krem;
  __shared__ unsigned c_gt, c_tie;
  unsigned prefix = 0;
  int krem = KT;
  for (int shift = 24; shift >= 0; shift -= 8) {
    hist[tid] = 0;
    __syncthreads();
    unsigned mask = (shift == 24) ? 0u : (0xFFFFFFFFu << (shift + 8));
    for (int i = tid; i < SS; i += 256) {
      unsigned u = __float_as_uint(v[i]);
      if ((u & mask) == (prefix & mask)) atomicAdd(&hist[(u >> shift) & 0xFFu], 1u);
    }
    __syncthreads();
    if (tid == 0) {
      int cum = 0, bsel = 0;
      for (int bin = 255; bin >= 0; --bin) {
        int c = (int)hist[bin];
        if (cum + c >= krem) { bsel = bin; break; }
        cum += c;
      }
      sh_pref = prefix | ((unsigned)bsel << shift);
      sh_krem = krem - cum;
    }
    __syncthreads();
    prefix = sh_pref;
    krem   = sh_krem;
    __syncthreads();
  }
  if (tid == 0) { c_gt = 0; c_tie = 0; }
  __syncthreads();
  int base_tie = KT - krem;
  for (int i = tid; i < SS; i += 256) {
    unsigned u = __float_as_uint(v[i]);
    if (u > prefix) {
      unsigned p = atomicAdd(&c_gt, 1u);
      idx[b * KT + (int)p] = i;
    } else if (u == prefix) {
      unsigned p = atomicAdd(&c_tie, 1u);
      if ((int)p < krem) idx[b * KT + base_tie + (int)p] = i;
    }
  }
}

// ---------------------------------------------------------------------------
// 4) gather tokens: x*imp at idx + sinusoidal positional encoding
// ---------------------------------------------------------------------------
__global__ __launch_bounds__(128)
void gather_kernel(const float* __restrict__ x, const float* __restrict__ imp,
                   const int* __restrict__ idx, float* __restrict__ tok32,
                   _Float16* __restrict__ tok16) {
  int row = blockIdx.x;               // b*KT + k
  int b   = row / KT;
  int tid = threadIdx.x;              // channel
  int s   = idx[row];
  float val = x[((size_t)b * CC + tid) * SS + s] * imp[(size_t)b * SS + s];
  int cp = tid & ~1;
  float dv  = __expf(-(float)cp * (9.210340371976184f / 128.f)); // ln(1e4)/dim
  float ang = (float)s * dv;
  float pos = (tid & 1) ? cosf(ang) : sinf(ang);
  float t = val + pos;
  tok32[(size_t)row * CC + tid] = t;
  tok16[(size_t)row * CC + tid] = (_Float16)t;
}

__global__ __launch_bounds__(256)
void f32_to_f16(const float* __restrict__ in, _Float16* __restrict__ out, int n) {
  int i = blockIdx.x * 256 + threadIdx.x;
  if (i < n) out[i] = (_Float16)in[i];
}

// ---------------------------------------------------------------------------
// Generic WMMA GEMM: out[m,n] = sum_k A16[m,k]*W16[n,k] + bias[n] (opt relu)
// A: [M,Kd] f16 row-major, W: [N,Kd] f16 row-major (== B^T), out f32 [M,N],
// optional f16 mirror. One wave per 16x64 tile; grid=(N/64, M/16), block=32.
// ---------------------------------------------------------------------------
__global__ __launch_bounds__(32)
void gemm_kernel(const _Float16* __restrict__ A, const _Float16* __restrict__ W,
                 const float* __restrict__ bias, float* __restrict__ out32,
                 _Float16* __restrict__ out16, int N, int Kd, int relu) {
  int lane = threadIdx.x, half_ = lane >> 4, lm = lane & 15;
  int row0 = blockIdx.y * 16, col0 = blockIdx.x * 64;
  v8f acc[4] = {v8f{}, v8f{}, v8f{}, v8f{}};
  const _Float16* arow = A + (size_t)(row0 + lm) * Kd;
  for (int k0 = 0; k0 < Kd; k0 += 32) {
    if (k0 + 32 < Kd) __builtin_prefetch(arow + k0 + 32, 0, 1);
    v16h af = load_a_frag(arow + k0, half_);
#pragma unroll
    for (int t = 0; t < 4; ++t) {
      v16h bf = load_b_frag(W + (size_t)(col0 + t * 16 + lm) * Kd + k0, half_);
      acc[t] = wmma_f16(af, bf, acc[t]);
    }
  }
#pragma unroll
  for (int t = 0; t < 4; ++t) {
    int col = col0 + t * 16 + lm;
    float bb = bias[col];
#pragma unroll
    for (int r = 0; r < 8; ++r) {
      float vv = acc[t][r] + bb;
      if (relu) vv = fmaxf(vv, 0.f);
      size_t o = (size_t)(row0 + half_ * 8 + r) * N + col;
      out32[o] = vv;
      if (out16) out16[o] = (_Float16)vv;
    }
  }
}

// ---------------------------------------------------------------------------
// 5) split qkv -> q,k packed [B,NH,KT,32] (hd zero-padded to 32), v^T [B,NH,16,KT]
// ---------------------------------------------------------------------------
__global__ __launch_bounds__(128)
void pack_qkv(const float* __restrict__ qkv, _Float16* __restrict__ qp,
              _Float16* __restrict__ kp, _Float16* __restrict__ vt) {
  int row = blockIdx.x;             // b*KT + k
  int b = row / KT, kk = row - b * KT;
  int tid = threadIdx.x;
  float qv = qkv[(size_t)row * 384 + tid];
  float kv = qkv[(size_t)row * 384 + 128 + tid];
  float vv = qkv[(size_t)row * 384 + 256 + tid];
  int h = tid >> 4, d = tid & 15, bh = b * NHH + h;
  size_t qo = ((size_t)bh * KT + kk) * 32 + d;
  qp[qo] = (_Float16)qv;  qp[qo + 16] = (_Float16)0.f;
  kp[qo] = (_Float16)kv;  kp[qo + 16] = (_Float16)0.f;
  vt[((size_t)bh * HD + d) * KT + kk] = (_Float16)vv;
}

// ---------------------------------------------------------------------------
// 6) flash-style attention: one wave per (b, h, 16-query tile)
// ---------------------------------------------------------------------------
__global__ __launch_bounds__(32)
void attn_kernel(const _Float16* __restrict__ qp, const _Float16* __restrict__ kp,
                 const _Float16* __restrict__ vt, _Float16* __restrict__ ocat) {
  int lane = threadIdx.x, half_ = lane >> 4, lm = lane & 15;
  int q0 = blockIdx.x * 16, h = blockIdx.y, b = blockIdx.z;
  int bh = b * NHH + h;
  __shared__ __attribute__((aligned(64))) _Float16 plds[16][32];

  v16h qf = load_a_frag(qp + ((size_t)bh * KT + q0 + lm) * 32, half_);
  const _Float16* kbase = kp + (size_t)bh * KT * 32;
  const _Float16* vbase = vt + ((size_t)bh * HD + lm) * KT;

  v8f oacc = v8f{};
  float m[8], l[8];
#pragma unroll
  for (int r = 0; r < 8; ++r) { m[r] = -__builtin_inff(); l[r] = 0.f; }

  for (int kt = 0; kt < KT / 32; ++kt) {
    v16h kf0 = load_b_frag(kbase + (size_t)(kt * 32 + lm) * 32, half_);
    v16h kf1 = load_b_frag(kbase + (size_t)(kt * 32 + 16 + lm) * 32, half_);
    v8f z = v8f{};
    v8f s0 = wmma_f16(qf, kf0, z);
    v8f s1 = wmma_f16(qf, kf1, z);
#pragma unroll
    for (int r = 0; r < 8; ++r) {
      float a0 = s0[r] * 0.25f;           // 1/sqrt(hd)
      float a1 = s1[r] * 0.25f;
      float t = fmaxf(a0, a1);
      t = fmaxf(t, __shfl_xor(t, 1));
      t = fmaxf(t, __shfl_xor(t, 2));
      t = fmaxf(t, __shfl_xor(t, 4));
      t = fmaxf(t, __shfl_xor(t, 8));     // row max within 16-lane half
      float mn   = fmaxf(m[r], t);
      float corr = __expf(m[r] - mn);
      float p0 = __expf(a0 - mn);
      float p1 = __expf(a1 - mn);
      float rs = p0 + p1;
      rs += __shfl_xor(rs, 1);
      rs += __shfl_xor(rs, 2);
      rs += __shfl_xor(rs, 4);
      rs += __shfl_xor(rs, 8);
      l[r] = l[r] * corr + rs;
      m[r] = mn;
      oacc[r] *= corr;
      plds[half_ * 8 + r][lm]      = (_Float16)p0;   // C-layout -> LDS
      plds[half_ * 8 + r][16 + lm] = (_Float16)p1;
    }
    __syncthreads();
    v16h pf = load_a_frag(&plds[lm][0], half_);      // back as A-fragment
    v16h vf = load_b_frag(vbase + kt * 32, half_);
    oacc = wmma_f16(pf, vf, oacc);
    __syncthreads();
  }
#pragma unroll
  for (int r = 0; r < 8; ++r) {
    float o = oacc[r] / l[r];
    ocat[((size_t)b * KT + q0 + half_ * 8 + r) * CC + h * HD + lm] = (_Float16)o;
  }
}

// ---------------------------------------------------------------------------
// 7) LayerNorm(a + b) * g + beta  (one block of 128 per row)
// ---------------------------------------------------------------------------
__global__ __launch_bounds__(128)
void ln_kernel(const float* __restrict__ a, const float* __restrict__ bres,
               const float* __restrict__ g, const float* __restrict__ be,
               float* __restrict__ out32, _Float16* __restrict__ out16) {
  int row = blockIdx.x, tid = threadIdx.x;
  float xv = a[(size_t)row * CC + tid] + bres[(size_t)row * CC + tid];
  float s = xv, s2 = xv * xv;
#pragma unroll
  for (int off = 16; off >= 1; off >>= 1) {
    s  += __shfl_xor(s, off);
    s2 += __shfl_xor(s2, off);
  }
  __shared__ float red[8];
  int wid = tid >> 5, lane = tid & 31;
  if (lane == 0) { red[wid] = s; red[4 + wid] = s2; }
  __syncthreads();
  float sum = red[0] + red[1] + red[2] + red[3];
  float sq  = red[4] + red[5] + red[6] + red[7];
  float mean = sum * (1.f / CC);
  float var  = sq * (1.f / CC) - mean * mean;
  float inv  = rsqrtf(var + 1e-5f);
  float o = (xv - mean) * inv * g[tid] + be[tid];
  out32[(size_t)row * CC + tid] = o;
  if (out16) out16[(size_t)row * CC + tid] = (_Float16)o;
}

// 8) scatter transformed tokens back
__global__ __launch_bounds__(128)
void scatter_kernel(const float* __restrict__ t2, const int* __restrict__ idx,
                    float* __restrict__ out) {
  int row = blockIdx.x, b = row / KT, tid = threadIdx.x;
  int s = idx[row];
  out[((size_t)b * CC + tid) * SS + s] = t2[(size_t)row * CC + tid];
}

// ---------------------------------------------------------------------------
extern "C" void kernel_launch(void* const* d_in, const int* in_sizes, int n_in,
                              void* d_out, int out_size, void* d_ws, size_t ws_size,
                              hipStream_t stream) {
  (void)in_sizes; (void)n_in; (void)out_size; (void)ws_size;
  const float* x      = (const float*)d_in[0];
  const float* conv_w = (const float*)d_in[1];
  const float* conv_b = (const float*)d_in[2];
  const float* bn_g   = (const float*)d_in[3];
  const float* bn_b   = (const float*)d_in[4];
  const float* bn_m   = (const float*)d_in[5];
  const float* bn_v   = (const float*)d_in[6];
  const float* w_qkv  = (const float*)d_in[7];
  const float* b_qkv  = (const float*)d_in[8];
  const float* w_o    = (const float*)d_in[9];
  const float* b_o    = (const float*)d_in[10];
  const float* ln1_g  = (const float*)d_in[11];
  const float* ln1_b  = (const float*)d_in[12];
  const float* w1     = (const float*)d_in[13];
  const float* b1     = (const float*)d_in[14];
  const float* w2     = (const float*)d_in[15];
  const float* b2     = (const float*)d_in[16];
  const float* ln2_g  = (const float*)d_in[17];
  const float* ln2_b  = (const float*)d_in[18];
  float* out = (float*)d_out;

  char* ws = (char*)d_ws;
  size_t off = 0;
  auto alloc = [&](size_t bytes) -> char* {
    char* p = ws + off;
    off = (off + bytes + 255) & ~(size_t)255;
    return p;
  };
  float*     imp    = (float*)alloc((size_t)BB * SS * 4);
  int*       idx    = (int*)alloc((size_t)BB * KT * 4);
  float*     tok32  = (float*)alloc((size_t)MR * CC * 4);
  _Float16*  tok16  = (_Float16*)alloc((size_t)MR * CC * 2);   // later: ocat
  float*     qkvbuf = (float*)alloc((size_t)MR * 384 * 4);     // later: h32+h16
  _Float16*  qp     = (_Float16*)alloc((size_t)BB * NHH * KT * 32 * 2); // later: t1_32
  _Float16*  kp     = (_Float16*)alloc((size_t)BB * NHH * KT * 32 * 2); // later: t1_16
  _Float16*  vt     = (_Float16*)alloc((size_t)BB * NHH * HD * KT * 2);
  float*     oproj  = (float*)alloc((size_t)MR * CC * 4);      // later: t2
  _Float16*  wq16   = (_Float16*)alloc(384 * 128 * 2);
  _Float16*  wo16   = (_Float16*)alloc(128 * 128 * 2);
  _Float16*  w116   = (_Float16*)alloc(256 * 128 * 2);
  _Float16*  w216   = (_Float16*)alloc(128 * 256 * 2);
  (void)vt;
  // lifetime-safe aliases
  _Float16* ocat = tok16;                                  // tok16 dead after QKV gemm
  float*    h32  = qkvbuf;                                 // qkv dead after pack
  _Float16* h16  = (_Float16*)((float*)qkvbuf + (size_t)MR * 256);
  float*    t1   = (float*)qp;                             // qp/kp dead after attention
  _Float16* t116 = kp;
  float*    ff   = tok32;                                  // tok32 dead after ln1
  float*    t2   = oproj;                                  // oproj dead after ln1

  // 1) importance map
  conv_imp_kernel<<<dim3((SS + 255) / 256, BB), 256, 0, stream>>>(
      x, conv_w, conv_b, bn_g, bn_b, bn_m, bn_v, imp);
  // 2) default output = x * imp
  size_t tot = (size_t)BB * CC * SS;
  mulimp_kernel<<<(unsigned)((tot + 255) / 256), 256, 0, stream>>>(x, imp, out);
  // 3) top-K selection
  topk_kernel<<<BB, 256, 0, stream>>>(imp, idx);
  // 4) gather + pos enc
  gather_kernel<<<MR, 128, 0, stream>>>(x, imp, idx, tok32, tok16);
  // weights -> f16
  f32_to_f16<<<(384 * 128 + 255) / 256, 256, 0, stream>>>(w_qkv, wq16, 384 * 128);
  f32_to_f16<<<(128 * 128 + 255) / 256, 256, 0, stream>>>(w_o, wo16, 128 * 128);
  f32_to_f16<<<(256 * 128 + 255) / 256, 256, 0, stream>>>(w1, w116, 256 * 128);
  f32_to_f16<<<(128 * 256 + 255) / 256, 256, 0, stream>>>(w2, w216, 128 * 256);
  // 5) QKV projection (WMMA)
  gemm_kernel<<<dim3(384 / 64, MR / 16), 32, 0, stream>>>(
      tok16, wq16, b_qkv, qkvbuf, (_Float16*)nullptr, 384, 128, 0);
  pack_qkv<<<MR, 128, 0, stream>>>(qkvbuf, qp, kp, vt);
  // 6) attention (WMMA, flash-style)
  attn_kernel<<<dim3(KT / 16, NHH, BB), 32, 0, stream>>>(qp, kp, vt, ocat);
  // out projection (WMMA)
  gemm_kernel<<<dim3(128 / 64, MR / 16), 32, 0, stream>>>(
      ocat, wo16, b_o, oproj, (_Float16*)nullptr, 128, 128, 0);
  // 7) LN1, FFN (WMMA), LN2
  ln_kernel<<<MR, 128, 0, stream>>>(tok32, oproj, ln1_g, ln1_b, t1, t116);
  gemm_kernel<<<dim3(256 / 64, MR / 16), 32, 0, stream>>>(
      t116, w116, b1, h32, h16, 256, 128, 1);
  gemm_kernel<<<dim3(128 / 64, MR / 16), 32, 0, stream>>>(
      h16, w216, b2, ff, (_Float16*)nullptr, 128, 256, 0);
  ln_kernel<<<MR, 128, 0, stream>>>(t1, ff, ln2_g, ln2_b, t2, (_Float16*)nullptr);
  // 8) scatter tokens back into the output map
  scatter_kernel<<<MR, 128, 0, stream>>>(t2, idx, out);
}
